// SetTransformer_37890201485321
// MI455X (gfx1250) — compile-verified, matrix operand order checked by reading, not run on hardware
//
#include <hip/hip_runtime.h>
#include <math.h>

// ---------------------------------------------------------------------------
// Set Transformer forward for MI455X (gfx1250, wave32, WMMA).
// f16 inputs / f32 accumulation via v_wmma_f32_16x16x32_f16.
// Software-pipelined: register staging + double-buffered LDS, 1 barrier/chunk.
// ---------------------------------------------------------------------------

typedef __attribute__((ext_vector_type(16))) _Float16 v16h;
typedef __attribute__((ext_vector_type(8)))  float    v8f;

#define EMBD   512
#define HEADS  8
#define HDIM   64
#define NSEEDS 4
#define BATCH  16
#define SEQ    1024

#define GF_RELU  1
#define GF_RESID 2

// K index for 16-bit A/B fragment pair p (0..7), half-select hsel (lane>>4):
// VGPR p holds K = base + hsel*8 + offsets per CDNA5 ISA 7.12.2 layout.
__device__ __forceinline__ int kmap(int p, int hsel) {
    return ((p >> 2) << 4) + (hsel << 3) + ((p & 3) << 1);
}

// convert float4 -> 4 packed halves, store as one 8-byte LDS write
__device__ __forceinline__ void st4h(_Float16* dst, float4 v) {
    union { _Float16 h[4]; uint2 u; } t;
    t.h[0] = (_Float16)v.x; t.h[1] = (_Float16)v.y;
    t.h[2] = (_Float16)v.z; t.h[3] = (_Float16)v.w;
    *(uint2*)dst = t.u;
}

// ---------------------------------------------------------------------------
// GEMM: C[M,N] = epilogue(A[M,K] @ W[K,N] + bias), f16 WMMA, f32 accumulate.
// Block 256 threads = 8 waves; tile 64(M) x 64(N); each wave a 16x32 strip;
// K-chunk 32, double-buffered.  Requires M%64==0, N%64==0, K%32==0.
// ---------------------------------------------------------------------------
__launch_bounds__(256)
__global__ void gemm_wmma(const float* __restrict__ A, const float* __restrict__ W,
                          const float* __restrict__ bias, const float* __restrict__ resid,
                          float* __restrict__ C, int M, int N, int K, int flags)
{
    __shared__ _Float16 As[2][64][40];   // padded stride: conflict-free gathers
    __shared__ _Float16 Bs[2][64][40];   // W chunk stored transposed: Bs[n][k]

    const int tid   = threadIdx.x;
    const int lane  = tid & 31;
    const int wave  = tid >> 5;
    const int m0    = blockIdx.y * 64;
    const int n0    = blockIdx.x * 64;
    const int tm    = (wave >> 1) * 16;   // wave's 16x32 strip
    const int tn    = (wave & 1)  * 32;
    const int row16 = lane & 15;
    const int hsel  = lane >> 4;

    // per-thread staging coordinates (2 float4 for A, 2 float4 for W)
    const int ar0 = tid >> 3,          ac0 = (tid & 7) * 4;
    const int ar1 = (tid + 256) >> 3,  ac1 = (tid & 7) * 4;     // rows 32..63
    const int wr0 = tid >> 4,          wc0 = (tid & 15) * 4;
    const int wr1 = (tid + 256) >> 4,  wc1 = (tid & 15) * 4;    // k rows 16..31

    // advancing global pointers (A: +32 floats, W: +32 rows per chunk)
    const float* aP0 = A + (size_t)(m0 + ar0) * K + ac0;
    const float* aP1 = A + (size_t)(m0 + ar1) * K + ac1;
    const float* wP0 = W + (size_t)wr0 * N + n0 + wc0;
    const float* wP1 = W + (size_t)wr1 * N + n0 + wc1;

    // prologue: chunk 0 into registers
    float4 ra0 = *(const float4*)aP0;
    float4 ra1 = *(const float4*)aP1;
    float4 rw0 = *(const float4*)wP0;
    float4 rw1 = *(const float4*)wP1;

    v8f acc0 = {}, acc1 = {};
    int buf = 0;

    for (int k0 = 0; k0 < K; k0 += 32) {
        // commit staged registers to LDS
        st4h(&As[buf][ar0][ac0], ra0);
        st4h(&As[buf][ar1][ac1], ra1);
        Bs[buf][wc0 + 0][wr0] = (_Float16)rw0.x;
        Bs[buf][wc0 + 1][wr0] = (_Float16)rw0.y;
        Bs[buf][wc0 + 2][wr0] = (_Float16)rw0.z;
        Bs[buf][wc0 + 3][wr0] = (_Float16)rw0.w;
        Bs[buf][wc1 + 0][wr1] = (_Float16)rw1.x;
        Bs[buf][wc1 + 1][wr1] = (_Float16)rw1.y;
        Bs[buf][wc1 + 2][wr1] = (_Float16)rw1.z;
        Bs[buf][wc1 + 3][wr1] = (_Float16)rw1.w;

        // issue next chunk's loads: latency hidden behind barrier + WMMA
        if (k0 + 32 < K) {
            aP0 += 32;  aP1 += 32;
            wP0 += (size_t)32 * N;  wP1 += (size_t)32 * N;
            ra0 = *(const float4*)aP0;
            ra1 = *(const float4*)aP1;
            rw0 = *(const float4*)wP0;
            rw1 = *(const float4*)wP1;
        }
        __syncthreads();

        v16h a, b0, b1;
        #pragma unroll
        for (int p = 0; p < 8; ++p) {
            int kk = kmap(p, hsel);
            a [2*p] = As[buf][tm + row16][kk];       a [2*p+1] = As[buf][tm + row16][kk + 1];
            b0[2*p] = Bs[buf][tn + row16][kk];       b0[2*p+1] = Bs[buf][tn + row16][kk + 1];
            b1[2*p] = Bs[buf][tn + 16 + row16][kk];  b1[2*p+1] = Bs[buf][tn + 16 + row16][kk + 1];
        }
        acc0 = __builtin_amdgcn_wmma_f32_16x16x32_f16(false, a, false, b0,
                                                      (short)0, acc0, false, false);
        acc1 = __builtin_amdgcn_wmma_f32_16x16x32_f16(false, a, false, b1,
                                                      (short)0, acc1, false, false);
        buf ^= 1;
    }

    // epilogue: bias (+relu) (+residual).  C layout: lane = n + 16*(m>>3), vgpr = m&7
    #pragma unroll
    for (int vv = 0; vv < 8; ++vv) {
        int    m    = vv + 8 * hsel;
        size_t base = (size_t)(m0 + tm + m) * N;
        int    gc0  = n0 + tn + row16;
        float v0 = acc0[vv] + bias[gc0];
        float v1 = acc1[vv] + bias[gc0 + 16];
        if (flags & GF_RELU)  { v0 = fmaxf(v0, 0.f);       v1 = fmaxf(v1, 0.f); }
        if (flags & GF_RESID) { v0 += resid[base + gc0];   v1 += resid[base + gc0 + 16]; }
        C[base + gc0]      = v0;
        C[base + gc0 + 16] = v1;
    }
}

// ---------------------------------------------------------------------------
// Flash attention per (batch, head, 64 q-rows). Block = 128 thr = 4 waves,
// each wave owns a 16-row Q tile.  out = q + softmax(mask(q k^T / 8)) v
// K/V chunks of 32 keys, register-pipelined + double-buffered LDS.
// ---------------------------------------------------------------------------
__launch_bounds__(128)
__global__ void attn_wmma(const float* __restrict__ q, const float* __restrict__ k,
                          const float* __restrict__ v, const float* __restrict__ mask,
                          float* __restrict__ out, int Nq, int Nk, int useMask)
{
    __shared__ _Float16 Qs[4][16][72];
    __shared__ _Float16 Ks[2][32][72];
    __shared__ _Float16 Vs[2][32][72];
    __shared__ _Float16 Ps[4][16][40];
    __shared__ float    smq[4][16];
    __shared__ float    smk[2][32];

    const int tid   = threadIdx.x;
    const int lane  = tid & 31;
    const int wave  = tid >> 5;
    const int head  = blockIdx.y;
    const int b     = blockIdx.z;
    const int q0    = (blockIdx.x * 4 + wave) * 16;
    const int row16 = lane & 15;
    const int hsel  = lane >> 4;
    const bool full = (q0 + 16 <= Nq);     // wave-uniform: no EXEC churn inside

    const size_t qbase = (size_t)b * Nq * EMBD + head * HDIM;
    const size_t kbase = (size_t)b * Nk * EMBD + head * HDIM;

    // stage this wave's Q tile (16x64 f32 = 256 float4), 8 float4 per lane
    if (full) {
        #pragma unroll
        for (int t = 0; t < 8; ++t) {
            int f = t * 32 + lane;
            int r = f >> 4, c4 = (f & 15) * 4;
            st4h(&Qs[wave][r][c4],
                 *(const float4*)(q + qbase + (size_t)(q0 + r) * EMBD + c4));
        }
    } else {
        #pragma unroll
        for (int t = 0; t < 8; ++t) {
            int f = t * 32 + lane;
            int r = f >> 4, c4 = (f & 15) * 4;
            float4 qv = make_float4(0.f, 0.f, 0.f, 0.f);
            if (q0 + r < Nq)
                qv = *(const float4*)(q + qbase + (size_t)(q0 + r) * EMBD + c4);
            st4h(&Qs[wave][r][c4], qv);
        }
    }
    if (lane < 16) {
        float mv = 0.f;
        if (useMask && (q0 + lane) < Nq) mv = mask[(size_t)b * Nq + q0 + lane];
        smq[wave][lane] = mv;
    }

    // per-thread K/V staging coords (4 float4 per array)
    int sr[4], sc[4];
    #pragma unroll
    for (int t = 0; t < 4; ++t) {
        int f = t * 128 + tid;
        sr[t] = f >> 4;
        sc[t] = (f & 15) * 4;
    }
    // prologue: chunk kb=0 into registers
    float4 rk[4], rv[4];
    #pragma unroll
    for (int t = 0; t < 4; ++t) {
        rk[t] = *(const float4*)(k + kbase + (size_t)sr[t] * EMBD + sc[t]);
        rv[t] = *(const float4*)(v + kbase + (size_t)sr[t] * EMBD + sc[t]);
    }
    float mreg = 0.f;
    if (useMask && wave == 0) mreg = mask[(size_t)b * Nk + lane];

    __syncthreads();

    // Q A-fragments: K=0..31 and K=32..63
    v16h aq0, aq1;
    #pragma unroll
    for (int p = 0; p < 8; ++p) {
        int kk = kmap(p, hsel);
        aq0[2*p]   = Qs[wave][row16][kk];
        aq0[2*p+1] = Qs[wave][row16][kk + 1];
        aq1[2*p]   = Qs[wave][row16][32 + kk];
        aq1[2*p+1] = Qs[wave][row16][32 + kk + 1];
    }

    float mi[8], li[8];
    #pragma unroll
    for (int i = 0; i < 8; ++i) { mi[i] = -1e30f; li[i] = 0.f; }
    v8f O0 = {}, O1 = {}, O2 = {}, O3 = {};
    int buf = 0;

    for (int kb = 0; kb < Nk; kb += 32) {
        // commit staged K/V registers to LDS
        #pragma unroll
        for (int t = 0; t < 4; ++t) {
            st4h(&Ks[buf][sr[t]][sc[t]], rk[t]);
            st4h(&Vs[buf][sr[t]][sc[t]], rv[t]);
        }
        if (useMask && wave == 0) smk[buf][lane] = mreg;

        // issue next chunk's loads (overlap barrier + softmax + WMMA)
        if (kb + 32 < Nk) {
            #pragma unroll
            for (int t = 0; t < 4; ++t) {
                rk[t] = *(const float4*)(k + kbase + (size_t)(kb + 32 + sr[t]) * EMBD + sc[t]);
                rv[t] = *(const float4*)(v + kbase + (size_t)(kb + 32 + sr[t]) * EMBD + sc[t]);
            }
            if (useMask && wave == 0) mreg = mask[(size_t)b * Nk + kb + 32 + lane];
        }
        __syncthreads();

        // S = Q @ K^T : two 16-key tiles, GEMM-K = 64 (2 chained WMMAs each)
        v16h bk;
        v8f S0 = {}, S1 = {};
        #pragma unroll
        for (int p = 0; p < 8; ++p) { int kk = kmap(p, hsel);
            bk[2*p] = Ks[buf][row16][kk]; bk[2*p+1] = Ks[buf][row16][kk + 1]; }
        S0 = __builtin_amdgcn_wmma_f32_16x16x32_f16(false, aq0, false, bk, (short)0, S0, false, false);
        #pragma unroll
        for (int p = 0; p < 8; ++p) { int kk = kmap(p, hsel);
            bk[2*p] = Ks[buf][row16][32 + kk]; bk[2*p+1] = Ks[buf][row16][32 + kk + 1]; }
        S0 = __builtin_amdgcn_wmma_f32_16x16x32_f16(false, aq1, false, bk, (short)0, S0, false, false);
        #pragma unroll
        for (int p = 0; p < 8; ++p) { int kk = kmap(p, hsel);
            bk[2*p] = Ks[buf][16 + row16][kk]; bk[2*p+1] = Ks[buf][16 + row16][kk + 1]; }
        S1 = __builtin_amdgcn_wmma_f32_16x16x32_f16(false, aq0, false, bk, (short)0, S1, false, false);
        #pragma unroll
        for (int p = 0; p < 8; ++p) { int kk = kmap(p, hsel);
            bk[2*p] = Ks[buf][16 + row16][32 + kk]; bk[2*p+1] = Ks[buf][16 + row16][32 + kk + 1]; }
        S1 = __builtin_amdgcn_wmma_f32_16x16x32_f16(false, aq1, false, bk, (short)0, S1, false, false);

        // scale, mask, online softmax (rows live in 16-lane halves)
        const float sc_ = 0.125f;                      // 1/sqrt(64)
        float km0 = useMask ? smk[buf][row16]      : 1.f;
        float km1 = useMask ? smk[buf][16 + row16] : 1.f;
        float P0[8], P1[8], alpha[8];
        #pragma unroll
        for (int vv = 0; vv < 8; ++vv) {
            float s0 = S0[vv] * sc_, s1 = S1[vv] * sc_;
            if (useMask) {
                float qm = smq[wave][vv + 8 * hsel];
                if (qm * km0 != 1.f) s0 = -1e9f;
                if (qm * km1 != 1.f) s1 = -1e9f;
            }
            float r = fmaxf(s0, s1);
            r = fmaxf(r, __shfl_xor(r, 1, 32));
            r = fmaxf(r, __shfl_xor(r, 2, 32));
            r = fmaxf(r, __shfl_xor(r, 4, 32));
            r = fmaxf(r, __shfl_xor(r, 8, 32));
            float mnew = fmaxf(mi[vv], r);
            alpha[vv] = __expf(mi[vv] - mnew);
            float p0 = __expf(s0 - mnew), p1 = __expf(s1 - mnew);
            float rs = p0 + p1;
            rs += __shfl_xor(rs, 1, 32);
            rs += __shfl_xor(rs, 2, 32);
            rs += __shfl_xor(rs, 4, 32);
            rs += __shfl_xor(rs, 8, 32);
            li[vv] = li[vv] * alpha[vv] + rs;
            mi[vv] = mnew;
            P0[vv] = p0; P1[vv] = p1;
        }

        // rescale O, stage P (C-layout -> LDS) for A-fragment conversion.
        // Per-wave private region; LDS ops are in-order per wave -> no barrier.
        #pragma unroll
        for (int vv = 0; vv < 8; ++vv) {
            O0[vv] *= alpha[vv]; O1[vv] *= alpha[vv];
            O2[vv] *= alpha[vv]; O3[vv] *= alpha[vv];
            int m = vv + 8 * hsel;
            Ps[wave][m][row16]      = (_Float16)P0[vv];
            Ps[wave][m][16 + row16] = (_Float16)P1[vv];
        }

        v16h ap;
        #pragma unroll
        for (int p = 0; p < 8; ++p) { int kk = kmap(p, hsel);
            ap[2*p] = Ps[wave][row16][kk]; ap[2*p+1] = Ps[wave][row16][kk + 1]; }

        // O += P @ V over the four 16-wide d tiles
        v16h bv;
        #pragma unroll
        for (int p = 0; p < 8; ++p) { int kk = kmap(p, hsel);
            bv[2*p] = Vs[buf][kk][row16]; bv[2*p+1] = Vs[buf][kk + 1][row16]; }
        O0 = __builtin_amdgcn_wmma_f32_16x16x32_f16(false, ap, false, bv, (short)0, O0, false, false);
        #pragma unroll
        for (int p = 0; p < 8; ++p) { int kk = kmap(p, hsel);
            bv[2*p] = Vs[buf][kk][16 + row16]; bv[2*p+1] = Vs[buf][kk + 1][16 + row16]; }
        O1 = __builtin_amdgcn_wmma_f32_16x16x32_f16(false, ap, false, bv, (short)0, O1, false, false);
        #pragma unroll
        for (int p = 0; p < 8; ++p) { int kk = kmap(p, hsel);
            bv[2*p] = Vs[buf][kk][32 + row16]; bv[2*p+1] = Vs[buf][kk + 1][32 + row16]; }
        O2 = __builtin_amdgcn_wmma_f32_16x16x32_f16(false, ap, false, bv, (short)0, O2, false, false);
        #pragma unroll
        for (int p = 0; p < 8; ++p) { int kk = kmap(p, hsel);
            bv[2*p] = Vs[buf][kk][48 + row16]; bv[2*p+1] = Vs[buf][kk + 1][48 + row16]; }
        O3 = __builtin_amdgcn_wmma_f32_16x16x32_f16(false, ap, false, bv, (short)0, O3, false, false);

        buf ^= 1;
    }

    // epilogue: normalize, add q residual (reference: H = q + A@v)
    if (full) {
        #pragma unroll
        for (int vv = 0; vv < 8; ++vv) {
            int m = vv + 8 * hsel;
            float inv = 1.f / li[vv];
            size_t ob = qbase + (size_t)(q0 + m) * EMBD;
            int c0 = row16;
            out[ob +      c0] = O0[vv] * inv + q[ob +      c0];
            out[ob + 16 + c0] = O1[vv] * inv + q[ob + 16 + c0];
            out[ob + 32 + c0] = O2[vv] * inv + q[ob + 32 + c0];
            out[ob + 48 + c0] = O3[vv] * inv + q[ob + 48 + c0];
        }
    } else {
        #pragma unroll
        for (int vv = 0; vv < 8; ++vv) {
            int m  = vv + 8 * hsel;
            int gr = q0 + m;
            if (gr < Nq) {
                float inv = 1.f / li[vv];
                size_t ob = qbase + (size_t)gr * EMBD;
                int c0 = row16;
                out[ob +      c0] = O0[vv] * inv + q[ob +      c0];
                out[ob + 16 + c0] = O1[vv] * inv + q[ob + 16 + c0];
                out[ob + 32 + c0] = O2[vv] * inv + q[ob + 32 + c0];
                out[ob + 48 + c0] = O3[vv] * inv + q[ob + 48 + c0];
            }
        }
    }
}

// ---------------------------------------------------------------------------
// Row-wise LayerNorm over 512 cols (+ optional ReLU). Block 128, 4 elems/thr.
// ---------------------------------------------------------------------------
__launch_bounds__(128)
__global__ void layernorm_rows(const float* __restrict__ in, float* __restrict__ out,
                               const float* __restrict__ g, const float* __restrict__ b,
                               int doRelu)
{
    __shared__ float red[128];
    const int row = blockIdx.x;
    const int tid = threadIdx.x;

    float4 xv = *(const float4*)(in + (size_t)row * EMBD + tid * 4);

    float s = xv.x + xv.y + xv.z + xv.w;
    red[tid] = s; __syncthreads();
    for (int off = 64; off > 0; off >>= 1) {
        if (tid < off) red[tid] += red[tid + off];
        __syncthreads();
    }
    float mu = red[0] * (1.f / EMBD);
    __syncthreads();

    float d0 = xv.x - mu, d1 = xv.y - mu, d2 = xv.z - mu, d3 = xv.w - mu;
    red[tid] = d0*d0 + d1*d1 + d2*d2 + d3*d3; __syncthreads();
    for (int off = 64; off > 0; off >>= 1) {
        if (tid < off) red[tid] += red[tid + off];
        __syncthreads();
    }
    float rstd = rsqrtf(red[0] * (1.f / EMBD) + 1e-5f);

    float4 gv = *(const float4*)(g + tid * 4);
    float4 bv = *(const float4*)(b + tid * 4);
    float4 y;
    y.x = d0 * rstd * gv.x + bv.x;
    y.y = d1 * rstd * gv.y + bv.y;
    y.z = d2 * rstd * gv.z + bv.z;
    y.w = d3 * rstd * gv.w + bv.w;
    if (doRelu) {
        y.x = fmaxf(y.x, 0.f); y.y = fmaxf(y.y, 0.f);
        y.z = fmaxf(y.z, 0.f); y.w = fmaxf(y.w, 0.f);
    }
    *(float4*)(out + (size_t)row * EMBD + tid * 4) = y;
}

// broadcast PMA seeds S[1,4,512] -> [16,4,512] (row r uses seed r&3)
__global__ void bcast_seeds(const float* __restrict__ S, float* __restrict__ out)
{
    int i = blockIdx.x * blockDim.x + threadIdx.x;   // 64*512 elems
    int r = i >> 9, c = i & 511;
    out[i] = S[(r & 3) * EMBD + c];
}

// ---------------------------------------------------------------------------
// Host orchestration
// ---------------------------------------------------------------------------
extern "C" void kernel_launch(void* const* d_in, const int* in_sizes, int n_in,
                              void* d_out, int out_size, void* d_ws, size_t ws_size,
                              hipStream_t stream)
{
    (void)in_sizes; (void)n_in; (void)out_size; (void)ws_size;

    const float* x    = (const float*)d_in[0];   // [16,1024,64]
    const float* mask = (const float*)d_in[1];   // [16,1024,1]
    const float* e_w  = (const float*)d_in[2];   // [64,512]
    const float* e_b  = (const float*)d_in[3];
    const float* e_g  = (const float*)d_in[4];
    const float* e_bb = (const float*)d_in[5];
    // sab0: d_in[6..17], sab1: d_in[18..29], pma.S: d_in[30], pma.mab: d_in[31..42]

    float* ws = (float*)d_ws;
    const size_t MN = (size_t)BATCH * SEQ * EMBD;   // 8M floats per buffer
    float* bufA = ws;                      // activations h
    float* bufQ = ws + MN;
    float* bufK = ws + 2 * MN;
    float* bufV = ws + 3 * MN;
    float* Sb   = ws + 4 * MN;             // [64,512] broadcast seeds
    float* qp   = Sb + 64 * EMBD;          // [64,512] PMA q
    float* Hp   = qp + 64 * EMBD;          // [64,512] PMA attn out
    float* Gp   = Hp + 64 * EMBD;          // [64,512] PMA fc_o out

    const int M = BATCH * SEQ;             // 16384
    const dim3 gemmGridBig(EMBD / 64, M / 64);
    const dim3 gemmGridSmall(EMBD / 64, 1);

    // 1) embed: relu(LN(x @ We + be))
    gemm_wmma<<<gemmGridBig, 256, 0, stream>>>(x, e_w, e_b, nullptr, bufQ, M, EMBD, 64, 0);
    layernorm_rows<<<M, 128, 0, stream>>>(bufQ, bufA, e_g, e_bb, 1);

    // 2) two SAB blocks (masked MAB(h, h))
    for (int s = 0; s < 2; ++s) {
        const float* const* p = (const float* const*)(d_in + 6 + s * 12);
        const float *Wq = p[0], *bq = p[1], *Wk = p[2], *bk = p[3];
        const float *Wv = p[4], *bv = p[5], *Wo = p[6], *bo = p[7];
        const float *g0 = p[8], *b0 = p[9], *g1 = p[10], *b1 = p[11];

        gemm_wmma<<<gemmGridBig, 256, 0, stream>>>(bufA, Wq, bq, nullptr, bufQ, M, EMBD, EMBD, 0);
        gemm_wmma<<<gemmGridBig, 256, 0, stream>>>(bufA, Wk, bk, nullptr, bufK, M, EMBD, EMBD, 0);
        gemm_wmma<<<gemmGridBig, 256, 0, stream>>>(bufA, Wv, bv, nullptr, bufV, M, EMBD, EMBD, 0);
        attn_wmma<<<dim3(SEQ / 64, HEADS, BATCH), 128, 0, stream>>>(
            bufQ, bufK, bufV, mask, bufA, SEQ, SEQ, 1);
        layernorm_rows<<<M, 128, 0, stream>>>(bufA, bufA, g0, b0, 0);
        gemm_wmma<<<gemmGridBig, 256, 0, stream>>>(bufA, Wo, bo, bufA, bufQ, M, EMBD, EMBD,
                                                   GF_RELU | GF_RESID);
        layernorm_rows<<<M, 128, 0, stream>>>(bufQ, bufA, g1, b1, 0);
    }

    // 3) PMA: MAB(S, h), no mask
    const float* S = (const float*)d_in[30];
    const float* const* p = (const float* const*)(d_in + 31);
    const float *Wq = p[0], *bq = p[1], *Wk = p[2], *bk = p[3];
    const float *Wv = p[4], *bv = p[5], *Wo = p[6], *bo = p[7];
    const float *g0 = p[8], *b0 = p[9], *g1 = p[10], *b1 = p[11];

    bcast_seeds<<<(64 * EMBD) / 256, 256, 0, stream>>>(S, Sb);
    gemm_wmma<<<gemmGridSmall, 256, 0, stream>>>(Sb,   Wq, bq, nullptr, qp,   64, EMBD, EMBD, 0);
    gemm_wmma<<<gemmGridBig,   256, 0, stream>>>(bufA, Wk, bk, nullptr, bufK, M,  EMBD, EMBD, 0);
    gemm_wmma<<<gemmGridBig,   256, 0, stream>>>(bufA, Wv, bv, nullptr, bufV, M,  EMBD, EMBD, 0);
    attn_wmma<<<dim3(1, HEADS, BATCH), 128, 0, stream>>>(
        qp, bufK, bufV, nullptr, Hp, NSEEDS, SEQ, 0);
    layernorm_rows<<<BATCH * NSEEDS, 128, 0, stream>>>(Hp, Hp, g0, b0, 0);
    gemm_wmma<<<gemmGridSmall, 256, 0, stream>>>(Hp, Wo, bo, Hp, Gp, BATCH * NSEEDS, EMBD, EMBD,
                                                 GF_RELU | GF_RESID);
    layernorm_rows<<<BATCH * NSEEDS, 128, 0, stream>>>(Gp, (float*)d_out, g1, b1, 0);
}
